// GATModuleNT_78262894067745
// MI455X (gfx1250) — compile-verified, matrix operand order checked by reading, not run on hardware
//
#include <hip/hip_runtime.h>
#include <hip/hip_bf16.h>

typedef __bf16 bf16_t;
typedef __attribute__((ext_vector_type(16))) __bf16 v16bf;
typedef __attribute__((ext_vector_type(8)))  float  v8f;

#define DEV __device__ __forceinline__

DEV bf16_t f2bf(float f) {
  unsigned u = __float_as_uint(f);
  unsigned r = u + 0x7FFFu + ((u >> 16) & 1u);   // round-to-nearest-even
  union { unsigned short s; bf16_t b; } cv;
  cv.s = (unsigned short)(r >> 16);
  return cv.b;
}

// order-preserving float <-> u32 encoding for atomicMax-based segment-max
DEV unsigned fenc(float f) {
  unsigned u = __float_as_uint(f);
  return (u & 0x80000000u) ? ~u : (u | 0x80000000u);
}
DEV float fdec(unsigned u) {
  return __uint_as_float((u & 0x80000000u) ? (u & 0x7FFFFFFFu) : ~u);
}

DEV float wave_sum(float v) {
  #pragma unroll
  for (int o = 16; o; o >>= 1) v += __shfl_xor(v, o, 32);
  return v;
}

// CDNA5 async global->LDS copy (ASYNCcnt tracked, no VGPR round-trip).
// lds_off = addrspace(3) byte offset (low 32 bits of the generic pointer).
DEV void async_ld_b128(unsigned lds_off, const bf16_t* g) {
  asm volatile("global_load_async_to_lds_b128 %0, %1, off"
               :: "v"(lds_off), "v"(g) : "memory");
}
#define S_WAIT_ASYNC(N) asm volatile("s_wait_asynccnt " #N ::: "memory")

// ---------------------------------------------------------------- fills
__global__ void k_fill_u32(unsigned* __restrict__ p, unsigned v, long long n) {
  long long i = (long long)blockIdx.x * blockDim.x + threadIdx.x;
  long long st = (long long)gridDim.x * blockDim.x;
  for (; i < n; i += st) p[i] = v;
}

// ---------------------------------------------------------------- input build
// x = [node_features | tag | density_feat | N_feat | zero-pad to 160] as bf16
__global__ void k_build_x(const float* __restrict__ nf, const int* __restrict__ pm,
                          const float* __restrict__ pf, bf16_t* __restrict__ xb,
                          int N, int F, int KP) {
  long long id = (long long)blockIdx.x * blockDim.x + threadIdx.x;
  if (id >= (long long)N * KP) return;
  int n = (int)(id / KP), c = (int)(id % KP);
  float v = 0.f;
  if (c < F) {
    v = nf[(long long)n * F + c];
  } else {
    int g = pm[n];
    bool solv = (n == N - 1) || (pm[n + 1] != g);
    if (c == F)          v = solv ? 1.f : 0.f;
    else if (c == F + 1) v = solv ? pf[g * 2 + 0] : 0.f;
    else if (c == F + 2) v = solv ? 0.f : pf[g * 2 + 1] * 0.1f;
  }
  xb[id] = f2bf(v);
}

// W (Korig x Nout, f32) -> Wt (Nout x Kpad, bf16), zero-padded K
__global__ void k_convert_wT(const float* __restrict__ W, bf16_t* __restrict__ Wt,
                             int Korig, int Kpad, int Nout) {
  long long id = (long long)blockIdx.x * blockDim.x + threadIdx.x;
  if (id >= (long long)Nout * Kpad) return;
  int n = (int)(id / Kpad), k = (int)(id % Kpad);
  Wt[id] = (k < Korig) ? f2bf(W[(long long)k * Nout + n]) : f2bf(0.f);
}

// ---------------------------------------------------------------- WMMA GEMM
// C(MxN,f32) = A(MxK bf16 row-major, K mult of 32) @ Bt(NxK bf16 row-major)^T
// 128x128 workgroup tile, 8 waves (4x2); each wave 32x64 via 2x4 wmma frags.
// Double-buffered LDS staging with GLOBAL_LOAD_ASYNC_TO_LDS_B128:
// every thread issues exactly 4 async b128 loads per tile, so
// s_wait_asynccnt 4 (in-order completion) pins the previous tile in LDS.
__global__ __launch_bounds__(256)
void k_gemm_bf16(const bf16_t* __restrict__ A, const bf16_t* __restrict__ Bt,
                 float* __restrict__ C, int M, int N, int K) {
  __shared__ alignas(16) bf16_t As[2][128][32];
  __shared__ alignas(16) bf16_t Bs[2][128][32];   // Bs[n][k]

  const int tid   = threadIdx.x;
  const int lane  = tid & 31;
  const int wid   = tid >> 5;
  const int waveM = wid >> 1, waveN = wid & 1;    // 4 x 2 waves
  const int half  = lane >> 4, l16 = lane & 15;
  const long long mBase = (long long)blockIdx.x * 128;
  const int nBase = blockIdx.y * 128;

  // staging: each 128x32 bf16 tile = 512 x 16B chunks; 2 A-chunks + 2 B-chunks
  // per thread -> uniform 4 async loads per thread per tile.
  const int r0 = tid >> 2;        // rows 0..63
  const int r1 = r0 + 64;         // rows 64..127
  const int sg = tid & 3;         // 16B segment within 64B row

  v8f acc[2][4];
  #pragma unroll
  for (int i = 0; i < 2; i++)
    #pragma unroll
    for (int j = 0; j < 4; j++)
      #pragma unroll
      for (int r = 0; r < 8; r++) acc[i][j][r] = 0.f;

  union FragCvt { struct { uint4 lo; uint4 hi; } q; v16bf v; };

  auto issue = [&](int buf, int k0) {
    async_ld_b128((unsigned)(size_t)&As[buf][r0][sg * 8],
                  A + (mBase + r0) * (long long)K + k0 + sg * 8);
    async_ld_b128((unsigned)(size_t)&As[buf][r1][sg * 8],
                  A + (mBase + r1) * (long long)K + k0 + sg * 8);
    async_ld_b128((unsigned)(size_t)&Bs[buf][r0][sg * 8],
                  Bt + (long long)(nBase + r0) * K + k0 + sg * 8);
    async_ld_b128((unsigned)(size_t)&Bs[buf][r1][sg * 8],
                  Bt + (long long)(nBase + r1) * K + k0 + sg * 8);
  };

  const int nk = K >> 5;
  issue(0, 0);
  for (int it = 0; it < nk; it++) {
    const int buf = it & 1;
    if (it + 1 < nk) {
      issue(buf ^ 1, (it + 1) << 5);
      S_WAIT_ASYNC(4);     // in-order: the 4 loads of tile `it` are in LDS
    } else {
      S_WAIT_ASYNC(0);
    }
    __syncthreads();

    v16bf afrag[2], bfrag[4];
    #pragma unroll
    for (int i = 0; i < 2; i++) {
      int m = waveM * 32 + i * 16 + l16;
      FragCvt f;
      f.q.lo = *(const uint4*)(&As[buf][m][half * 8]);
      f.q.hi = *(const uint4*)(&As[buf][m][16 + half * 8]);
      afrag[i] = f.v;
    }
    #pragma unroll
    for (int j = 0; j < 4; j++) {
      int n = waveN * 64 + j * 16 + l16;
      FragCvt f;
      f.q.lo = *(const uint4*)(&Bs[buf][n][half * 8]);
      f.q.hi = *(const uint4*)(&Bs[buf][n][16 + half * 8]);
      bfrag[j] = f.v;
    }
    #pragma unroll
    for (int i = 0; i < 2; i++)
      #pragma unroll
      for (int j = 0; j < 4; j++)
        acc[i][j] = __builtin_amdgcn_wmma_f32_16x16x32_bf16(
            false, afrag[i], false, bfrag[j], (short)0, acc[i][j], false, false);
    __syncthreads();   // all waves done reading `buf` before it is re-issued
  }

  #pragma unroll
  for (int i = 0; i < 2; i++)
    #pragma unroll
    for (int j = 0; j < 4; j++) {
      int col = nBase + waveN * 64 + j * 16 + l16;
      #pragma unroll
      for (int r = 0; r < 8; r++) {
        long long row = mBase + waveM * 32 + i * 16 + (half ? r + 8 : r);
        C[row * (long long)N + col] = acc[i][j][r];
      }
    }
}

// ---------------------------------------------------------------- attention
__global__ void k_node_att(const float* __restrict__ h, const float* __restrict__ att_s,
                           const float* __restrict__ att_d, float* __restrict__ asrc,
                           float* __restrict__ adst, int H, int C) {
  int n = blockIdx.x;
  int hh = threadIdx.x >> 5, lane = threadIdx.x & 31;
  const float4* hp = (const float4*)(h + (long long)n * H * C + hh * C);
  const float4* sp = (const float4*)(att_s + hh * C);
  const float4* dp = (const float4*)(att_d + hh * C);
  float4 hv = hp[lane], sv = sp[lane], dv = dp[lane];
  float ss = hv.x * sv.x + hv.y * sv.y + hv.z * sv.z + hv.w * sv.w;
  float dd = hv.x * dv.x + hv.y * dv.y + hv.z * dv.z + hv.w * dv.w;
  ss = wave_sum(ss); dd = wave_sum(dd);
  if (lane == 0) { asrc[(long long)n * H + hh] = ss; adst[(long long)n * H + hh] = dd; }
}

// ve[k*H+h] = sum_c We[k, h*C+c] * att_e[h,c]   (edge score is linear in ea)
__global__ void k_ve(const float* __restrict__ We, const float* __restrict__ att_e,
                     float* __restrict__ ve, int H, int C) {
  int t = threadIdx.x;
  if (t >= 2 * H) return;
  int k = t & 1, hh = t >> 1;
  float s = 0.f;
  for (int c = 0; c < C; c++)
    s += We[(long long)k * H * C + hh * C + c] * att_e[hh * C + c];
  ve[k * H + hh] = s;
}

__global__ void k_edge_sum(const float* __restrict__ ea, float* __restrict__ easum, int E) {
  float s0 = 0.f, s1 = 0.f;
  for (long long i = (long long)blockIdx.x * blockDim.x + threadIdx.x; i < E;
       i += (long long)gridDim.x * blockDim.x) {
    s0 += ea[2 * i]; s1 += ea[2 * i + 1];
  }
  s0 = wave_sum(s0); s1 = wave_sum(s1);
  if ((threadIdx.x & 31) == 0) { atomicAdd(&easum[0], s0); atomicAdd(&easum[1], s1); }
}

__global__ void k_alpha(const int* __restrict__ src, const int* __restrict__ dst,
                        const float* __restrict__ ea, const float* __restrict__ easum,
                        const float* __restrict__ ve, const float* __restrict__ asrc,
                        const float* __restrict__ adst, float* __restrict__ alpha,
                        unsigned* __restrict__ amax, int E, int N, int H) {
  long long t = (long long)blockIdx.x * blockDim.x + threadIdx.x;
  long long S = (long long)E + N;
  if (t >= S * H) return;
  int h = (int)(t % H);
  long long e = t / H;
  int s, d; float e0, e1;
  if (e < E) { s = src[e]; d = dst[e]; e0 = ea[2 * e]; e1 = ea[2 * e + 1]; }
  else { s = d = (int)(e - E); float inv = 1.0f / (float)E; e0 = easum[0] * inv; e1 = easum[1] * inv; }
  float a = asrc[(long long)s * H + h] + adst[(long long)d * H + h] + e0 * ve[h] + e1 * ve[H + h];
  float al = a > 0.f ? a : 0.2f * a;   // leaky relu
  alpha[t] = al;
  atomicMax(&amax[(long long)d * H + h], fenc(al));
}

__global__ void k_ex(const int* __restrict__ dst, const float* __restrict__ alpha,
                     const unsigned* __restrict__ amax, float* __restrict__ ex,
                     float* __restrict__ denom, int E, int N, int H) {
  long long t = (long long)blockIdx.x * blockDim.x + threadIdx.x;
  long long S = (long long)E + N;
  if (t >= S * H) return;
  int h = (int)(t % H);
  long long e = t / H;
  int d = (e < E) ? dst[e] : (int)(e - E);
  float m = fdec(amax[(long long)d * H + h]);
  float x = __expf(alpha[t] - m);
  ex[t] = x;
  atomicAdd(&denom[(long long)d * H + h], x);
}

__global__ void k_scatter(const int* __restrict__ src, const int* __restrict__ dst,
                          const float* __restrict__ h, const float* __restrict__ ex,
                          const float* __restrict__ denom, float* __restrict__ out,
                          int E, int N, int H, int C) {
  long long id = (long long)blockIdx.x * blockDim.x + threadIdx.x;
  long long S = (long long)E + N;
  long long HC = (long long)H * C;
  if (id >= S * HC) return;
  int c = (int)(id % C);
  long long eh = id / C;
  int hh = (int)(eh % H);
  long long e = eh / H;
  int s, d;
  if (e < E) { s = src[e]; d = dst[e]; } else { s = d = (int)(e - E); }
  long long t = e * H + hh;
  float attn = ex[t] / (denom[(long long)d * H + hh] + 1e-16f);
  atomicAdd(&out[(long long)d * HC + hh * C + c],
            h[(long long)s * HC + hh * C + c] * attn);
}

__global__ void k_bias_relu_bf16(const float* __restrict__ in, const float* __restrict__ b,
                                 bf16_t* __restrict__ outb, long long n, int J) {
  long long id = (long long)blockIdx.x * blockDim.x + threadIdx.x;
  if (id >= n) return;
  float v = in[id] + b[(int)(id % J)];
  outb[id] = f2bf(v > 0.f ? v : 0.f);
}

// ---------------------------------------------------------------- pooling
__global__ void k_pool_accum(const float* __restrict__ x, const float* __restrict__ b2,
                             const int* __restrict__ pm, float* __restrict__ msum,
                             float* __restrict__ ssum, int N, int C) {
  long long id = (long long)blockIdx.x * blockDim.x + threadIdx.x;
  if (id >= (long long)N * C) return;
  int n = (int)(id / C), c = (int)(id % C);
  int g = pm[n];
  bool solv = (n == N - 1) || (pm[n + 1] != g);
  float v = x[id] + b2[c];
  atomicAdd(&(solv ? ssum : msum)[(long long)g * C + c], v);
}

__global__ void k_pool_count(const int* __restrict__ pm, float* __restrict__ mcnt,
                             float* __restrict__ scnt, int N) {
  int n = blockIdx.x * blockDim.x + threadIdx.x;
  if (n >= N) return;
  int g = pm[n];
  bool solv = (n == N - 1) || (pm[n + 1] != g);
  atomicAdd(solv ? &scnt[g] : &mcnt[g], 1.0f);
}

__global__ void k_pool_final(const float* __restrict__ msum, const float* __restrict__ ssum,
                             const float* __restrict__ mcnt, const float* __restrict__ scnt,
                             float* __restrict__ out, int G, int C) {
  long long id = (long long)blockIdx.x * blockDim.x + threadIdx.x;
  if (id >= (long long)G * C) return;
  int g = (int)(id / C), c = (int)(id % C);
  out[(long long)g * 2 * C + c]     = msum[id] / fmaxf(mcnt[g], 1.f);
  out[(long long)g * 2 * C + C + c] = ssum[id] / fmaxf(scnt[g], 1.f);
}

// ---------------------------------------------------------------- host
extern "C" void kernel_launch(void* const* d_in, const int* in_sizes, int n_in,
                              void* d_out, int out_size, void* d_ws, size_t ws_size,
                              hipStream_t stream) {
  (void)n_in; (void)out_size; (void)ws_size;
  const float* nf  = (const float*)d_in[0];
  const int*   ei  = (const int*)  d_in[1];
  const float* ea  = (const float*)d_in[2];
  const int*   pm  = (const int*)  d_in[3];
  const float* pf  = (const float*)d_in[4];
  const float* W1  = (const float*)d_in[5];
  const float* as1 = (const float*)d_in[6];
  const float* ad1 = (const float*)d_in[7];
  const float* We1 = (const float*)d_in[8];
  const float* ae1 = (const float*)d_in[9];
  const float* b1  = (const float*)d_in[10];
  const float* W2  = (const float*)d_in[11];
  const float* as2 = (const float*)d_in[12];
  const float* ad2 = (const float*)d_in[13];
  const float* We2 = (const float*)d_in[14];
  const float* ae2 = (const float*)d_in[15];
  const float* b2  = (const float*)d_in[16];

  const int F = 128, H1 = 4, C = 128, HC1 = 512, H2 = 1;
  const int N = in_sizes[0] / F;
  const int E = in_sizes[2] / 2;
  const int G = in_sizes[4] / 2;
  const int K1 = F + 3, K1P = 160;      // pad K 131 -> 160 (mult of 32)
  const long long S = (long long)E + N;

  char* ws = (char*)d_ws;
  size_t off = 0;
  auto alloc = [&](size_t bytes) -> char* {
    off = (off + 255) & ~(size_t)255;
    char* p = ws + off; off += bytes; return p;
  };
  bf16_t*   xb    = (bf16_t*)  alloc((size_t)N * K1P * 2);
  bf16_t*   w1b   = (bf16_t*)  alloc((size_t)HC1 * K1P * 2);   // transposed
  float*    h1    = (float*)   alloc((size_t)N * HC1 * 4);
  float*    asrc1 = (float*)   alloc((size_t)N * H1 * 4);
  float*    adst1 = (float*)   alloc((size_t)N * H1 * 4);
  float*    ve1   = (float*)   alloc(2 * H1 * 4);
  float*    easum = (float*)   alloc(2 * 4);
  unsigned* amax1 = (unsigned*)alloc((size_t)N * H1 * 4);
  float*    den1  = (float*)   alloc((size_t)N * H1 * 4);
  float*    alp1  = (float*)   alloc((size_t)S * H1 * 4);
  float*    ex1   = (float*)   alloc((size_t)S * H1 * 4);
  float*    out1  = (float*)   alloc((size_t)N * HC1 * 4);
  bf16_t*   x2b   = (bf16_t*)  alloc((size_t)N * HC1 * 2);
  bf16_t*   w2b   = (bf16_t*)  alloc((size_t)C * HC1 * 2);     // transposed
  float*    asrc2 = (float*)   alloc((size_t)N * 4);
  float*    adst2 = (float*)   alloc((size_t)N * 4);
  float*    ve2   = (float*)   alloc(2 * 4);
  unsigned* amax2 = (unsigned*)alloc((size_t)N * 4);
  float*    den2  = (float*)   alloc((size_t)N * 4);
  float*    alp2  = (float*)   alloc((size_t)S * 4);
  float*    ex2   = (float*)   alloc((size_t)S * 4);
  float*    msum  = (float*)   alloc((size_t)G * C * 4);
  float*    ssum  = (float*)   alloc((size_t)G * C * 4);
  float*    mcnt  = (float*)   alloc((size_t)G * 4);
  float*    scnt  = (float*)   alloc((size_t)G * 4);
  float* h2   = h1;    // h1 dead after layer-1 scatter
  float* out2 = out1;  // out1 dead after bias_relu -> x2b

  const int B = 256;
  auto blocks = [](long long n, int b) { return (unsigned)((n + b - 1) / b); };

  // zero accumulators (every call: graph-replay determinism)
  k_fill_u32<<<2048, B, 0, stream>>>((unsigned*)out1, 0u, (long long)N * HC1);
  k_fill_u32<<<256,  B, 0, stream>>>(amax1, 0u, (long long)N * H1);
  k_fill_u32<<<256,  B, 0, stream>>>((unsigned*)den1, 0u, (long long)N * H1);
  k_fill_u32<<<1,   32, 0, stream>>>((unsigned*)easum, 0u, 2);
  k_fill_u32<<<256,  B, 0, stream>>>((unsigned*)msum, 0u, (long long)G * C);
  k_fill_u32<<<256,  B, 0, stream>>>((unsigned*)ssum, 0u, (long long)G * C);
  k_fill_u32<<<32,   B, 0, stream>>>((unsigned*)mcnt, 0u, G);
  k_fill_u32<<<32,   B, 0, stream>>>((unsigned*)scnt, 0u, G);

  // ---- layer 1
  k_build_x   <<<blocks((long long)N * K1P, B), B, 0, stream>>>(nf, pm, pf, xb, N, F, K1P);
  k_convert_wT<<<blocks((long long)HC1 * K1P, B), B, 0, stream>>>(W1, w1b, K1, K1P, HC1);
  { dim3 g(N / 128, HC1 / 128);
    k_gemm_bf16<<<g, 256, 0, stream>>>(xb, w1b, h1, N, HC1, K1P); }

  k_edge_sum<<<512, B, 0, stream>>>(ea, easum, E);
  k_ve      <<<1, 64, 0, stream>>>(We1, ae1, ve1, H1, C);
  k_node_att<<<N, 32 * H1, 0, stream>>>(h1, as1, ad1, asrc1, adst1, H1, C);
  k_alpha  <<<blocks(S * H1, B), B, 0, stream>>>(ei, ei + E, ea, easum, ve1, asrc1, adst1,
                                                 alp1, amax1, E, N, H1);
  k_ex     <<<blocks(S * H1, B), B, 0, stream>>>(ei + E, alp1, amax1, ex1, den1, E, N, H1);
  k_scatter<<<blocks(S * (long long)H1 * C, B), B, 0, stream>>>(ei, ei + E, h1, ex1, den1,
                                                                out1, E, N, H1, C);
  k_bias_relu_bf16<<<blocks((long long)N * HC1, B), B, 0, stream>>>(out1, b1, x2b,
                                                                    (long long)N * HC1, HC1);

  // ---- layer 2 (out2/amax2/den2 zeroed only after out1 consumed)
  k_fill_u32<<<1024, B, 0, stream>>>((unsigned*)out2, 0u, (long long)N * C);
  k_fill_u32<<<64,   B, 0, stream>>>(amax2, 0u, N);
  k_fill_u32<<<64,   B, 0, stream>>>((unsigned*)den2, 0u, N);

  k_convert_wT<<<blocks((long long)C * HC1, B), B, 0, stream>>>(W2, w2b, HC1, HC1, C);
  { dim3 g(N / 128, C / 128);
    k_gemm_bf16<<<g, 256, 0, stream>>>(x2b, w2b, h2, N, C, HC1); }

  k_ve      <<<1, 64, 0, stream>>>(We2, ae2, ve2, H2, C);
  k_node_att<<<N, 32 * H2, 0, stream>>>(h2, as2, ad2, asrc2, adst2, H2, C);
  k_alpha  <<<blocks(S * H2, B), B, 0, stream>>>(ei, ei + E, ea, easum, ve2, asrc2, adst2,
                                                 alp2, amax2, E, N, H2);
  k_ex     <<<blocks(S * H2, B), B, 0, stream>>>(ei + E, alp2, amax2, ex2, den2, E, N, H2);
  k_scatter<<<blocks(S * (long long)H2 * C, B), B, 0, stream>>>(ei, ei + E, h2, ex2, den2,
                                                                out2, E, N, H2, C);

  // ---- pooling (bias b2 folded in)
  k_pool_accum<<<blocks((long long)N * C, B), B, 0, stream>>>(out2, b2, pm, msum, ssum, N, C);
  k_pool_count<<<blocks(N, B), B, 0, stream>>>(pm, mcnt, scnt, N);
  k_pool_final<<<blocks((long long)G * C, B), B, 0, stream>>>(msum, ssum, mcnt, scnt,
                                                              (float*)d_out, G, C);
}
